// LlamaBlock_7507602833801
// MI455X (gfx1250) — matrix operand order repaired and checked by reading, not compiled, not run on hardware
//
#include <hip/hip_runtime.h>
#include <hip/hip_bf16.h>

// ---------------------------------------------------------------------------
// LLaMA block for MI455X (gfx1250). Compute-bound (~650 GFLOP vs ~400MB
// traffic) -> everything through v_wmma_f32_16x16x32_bf16, f32 accumulate.
// Weights are cast+transposed once so ALL hot-loop LDS staging is contiguous
// 16B/lane row copies via global_load_async_to_lds_b128 (ASYNCcnt tracked).
// ---------------------------------------------------------------------------

typedef __bf16 bf16;
typedef __attribute__((ext_vector_type(16))) __bf16 bf16x16;
typedef __attribute__((ext_vector_type(8)))  __bf16 bf16x8;
typedef __attribute__((ext_vector_type(8)))  float  f32x8;
typedef __attribute__((ext_vector_type(4)))  int    v4i;

#if defined(__has_builtin)
#if __has_builtin(__builtin_amdgcn_global_load_async_to_lds_b128)
#define HAVE_ASYNC_LDS 1
#else
#define HAVE_ASYNC_LDS 0
#endif
#else
#define HAVE_ASYNC_LDS 0
#endif

#if HAVE_ASYNC_LDS
typedef __attribute__((address_space(1))) v4i gv4i;  // global
typedef __attribute__((address_space(3))) v4i lv4i;  // LDS
#endif

__device__ __forceinline__ float bf2f(bf16 v) {
  unsigned short u = __builtin_bit_cast(unsigned short, v);
  unsigned int x = ((unsigned int)u) << 16;
  return __builtin_bit_cast(float, x);
}
__device__ __forceinline__ bf16 f2bf(float f) {
  unsigned int u = __builtin_bit_cast(unsigned int, f);
  u += 0x7FFFu + ((u >> 16) & 1u);
  unsigned short h = (unsigned short)(u >> 16);
  return __builtin_bit_cast(bf16, h);
}
__device__ __forceinline__ f32x8 zero8() {
  f32x8 z = {0.f, 0.f, 0.f, 0.f, 0.f, 0.f, 0.f, 0.f};
  return z;
}
__device__ __forceinline__ bf16x16 cat8(bf16x8 lo, bf16x8 hi) {
  return __builtin_shufflevector(lo, hi, 0, 1, 2, 3, 4, 5, 6, 7,
                                         8, 9, 10, 11, 12, 13, 14, 15);
}
__device__ __forceinline__ f32x8 wmma_bf16(bf16x16 a, bf16x16 b, f32x8 c) {
  return __builtin_amdgcn_wmma_f32_16x16x32_bf16(false, a, false, b,
                                                 (short)0, c, false, false);
}

// 16B/lane global -> LDS copy; async path when toolchain provides it.
__device__ __forceinline__ void copy16_to_lds(const bf16* g, bf16* l) {
#if HAVE_ASYNC_LDS
  __builtin_amdgcn_global_load_async_to_lds_b128((gv4i*)g, (lv4i*)l, 0, 0);
#else
  *(uint4*)l = *(const uint4*)g;
#endif
}
__device__ __forceinline__ void wait_async_lds() {
#if HAVE_ASYNC_LDS
#if __has_builtin(__builtin_amdgcn_s_wait_asynccnt)
  __builtin_amdgcn_s_wait_asynccnt(0);
#else
  asm volatile("s_wait_asynccnt 0x0" ::: "memory");
#endif
#endif
}

// ---------------------------------------------------------------------------
// Fused cast(f32->bf16) + transpose: out[n][k] = in[k][n]. 32x32 tiles.
// ---------------------------------------------------------------------------
__global__ __launch_bounds__(256) void cast_transpose_kernel(
    const float* __restrict__ in, bf16* __restrict__ out, int K, int N) {
  __shared__ bf16 tile[32][33];
  const int n0 = blockIdx.x * 32;
  const int k0 = blockIdx.y * 32;
  const int tx = threadIdx.x & 31;
  const int ty = threadIdx.x >> 5;  // 8 rows per pass
#pragma unroll
  for (int r = ty; r < 32; r += 8)
    tile[r][tx] = f2bf(in[(long long)(k0 + r) * N + n0 + tx]);
  __syncthreads();
#pragma unroll
  for (int r = ty; r < 32; r += 8)
    out[(long long)(n0 + r) * K + k0 + tx] = tile[tx][r];
}

// ---------------------------------------------------------------------------
// RMSNorm: one block (256 threads) per row of D=2048. f32 in -> bf16 out.
// ---------------------------------------------------------------------------
__global__ __launch_bounds__(256) void rmsnorm_kernel(
    const float* __restrict__ x, const float* __restrict__ w,
    bf16* __restrict__ out, int D) {
  const int row = blockIdx.x;
  const float* xr = x + (long long)row * D;
  float ss = 0.f;
  for (int i = threadIdx.x; i < D; i += 256) {
    float v = xr[i];
    ss += v * v;
  }
#pragma unroll
  for (int off = 1; off < 32; off <<= 1) ss += __shfl_xor(ss, off, 32);
  __shared__ float partial[8];
  if ((threadIdx.x & 31) == 0) partial[threadIdx.x >> 5] = ss;
  __syncthreads();
  float tot = 0.f;
#pragma unroll
  for (int i = 0; i < 8; i++) tot += partial[i];
  float inv = rsqrtf(tot / (float)D + 1e-6f);
  bf16* orow = out + (long long)row * D;
  for (int i = threadIdx.x; i < D; i += 256)
    orow[i] = f2bf(xr[i] * inv * w[i]);
}

// ---------------------------------------------------------------------------
// RoPE (in-place on bf16 [B*S, H*HD]); replicates reference's (i-1) exponent.
// ---------------------------------------------------------------------------
__global__ __launch_bounds__(256) void rope_kernel(bf16* __restrict__ X) {
  constexpr int S = 2048, D = 2048, H = 16, HD = 128, HHD = HD / 2;
  long long idx = (long long)blockIdx.x * blockDim.x + threadIdx.x;
  int i = (int)(idx & (HHD - 1));
  int h = (int)((idx >> 6) & (H - 1));
  int s = (int)((idx >> 10) & (S - 1));
  int b = (int)(idx >> 21);
  long long base = ((long long)(b * S + s)) * D + h * HD;
  float t = powf(10000.0f, -2.0f * ((float)i - 1.0f) / (float)HD);
  float ang = (float)s * t;
  float c = cosf(ang), sn = sinf(ang);
  float f = bf2f(X[base + i]);
  float s2 = bf2f(X[base + i + HHD]);
  X[base + i] = f2bf(f * c + s2 * sn);
  X[base + i + HHD] = f2bf(f * sn - s2 * c);
}

// ---------------------------------------------------------------------------
// Tiled bf16 WMMA GEMM: C[M,N] = A[M,K] @ Wt[N,K]^T + bias.
// Both operands row-major-in-K -> all LDS staging is contiguous row copies.
// Block 256 thr (8 waves), tile 128x128, K-step 64, double-buffered LDS.
// Wave tile 32x64 -> 2x4 = 8 accumulators of 16x16.
// ---------------------------------------------------------------------------
enum : int { EPI_BF16 = 0, EPI_RESID_F32 = 1, EPI_SILU_MUL = 2, EPI_VT = 3 };

#define BM 128
#define BN 128
#define BK 64

template <int EPI>
__global__ __launch_bounds__(256) void gemm_bf16_kernel(
    const bf16* __restrict__ A, const bf16* __restrict__ Wt,
    const float* __restrict__ bias, void* __restrict__ outv,
    const void* __restrict__ extrav, int M, int N, int K) {
  __shared__ __align__(16) bf16 As[2][BM][BK];
  __shared__ __align__(16) bf16 Bs[2][BN][BK];
  const int tid = threadIdx.x;
  const int wid = tid >> 5;
  const int lane = tid & 31;
  const int lhalf = lane >> 4;
  const int l16 = lane & 15;
  const int n0 = blockIdx.x * BN;
  const int m0 = blockIdx.y * BM;
  const int wm = (wid & 3) * 32;
  const int wn = (wid >> 2) * 64;

  f32x8 acc[2][4];
#pragma unroll
  for (int a = 0; a < 2; a++)
#pragma unroll
    for (int b = 0; b < 4; b++) acc[a][b] = zero8();

  // Stage one K-tile (BM/BN rows x BK) : 1024 chunks of 16B each side.
  auto load_tile = [&](int buf, int k0) {
#pragma unroll
    for (int i = tid; i < BM * (BK / 8); i += 256) {
      int r = i >> 3;
      int c = (i & 7) * 8;
      copy16_to_lds(A + (long long)(m0 + r) * K + k0 + c, &As[buf][r][c]);
    }
#pragma unroll
    for (int i = tid; i < BN * (BK / 8); i += 256) {
      int r = i >> 3;
      int c = (i & 7) * 8;
      copy16_to_lds(Wt + (long long)(n0 + r) * K + k0 + c, &Bs[buf][r][c]);
    }
  };

  const int KT = K / BK;
  load_tile(0, 0);
  wait_async_lds();
  __syncthreads();

  for (int t = 0; t < KT; t++) {
    const int buf = t & 1;
    if (t + 1 < KT) load_tile(buf ^ 1, (t + 1) * BK);

#pragma unroll
    for (int ks = 0; ks < 2; ks++) {
      bf16x16 afrag[2], bfrag[4];
      const int a0 = ks * 32 + lhalf * 8;
#pragma unroll
      for (int mt = 0; mt < 2; mt++) {
        const bf16* ap = &As[buf][wm + mt * 16 + l16][0];
        afrag[mt] = cat8(*(const bf16x8*)(ap + a0),
                         *(const bf16x8*)(ap + a0 + 16));
      }
      const int b0 = ks * 32 + lhalf * 8;
#pragma unroll
      for (int nt = 0; nt < 4; nt++) {
        const bf16* bp = &Bs[buf][wn + nt * 16 + l16][0];
        bfrag[nt] = cat8(*(const bf16x8*)(bp + b0),
                         *(const bf16x8*)(bp + b0 + 16));
      }
#pragma unroll
      for (int mt = 0; mt < 2; mt++)
#pragma unroll
        for (int nt = 0; nt < 4; nt++)
          acc[mt][nt] = wmma_bf16(afrag[mt], bfrag[nt], acc[mt][nt]);
    }
    wait_async_lds();
    __syncthreads();
  }

  // Epilogue. C layout: VGPR r -> row r (+8 for lanes>=16), col = l16.
#pragma unroll
  for (int mt = 0; mt < 2; mt++) {
#pragma unroll
    for (int nt = 0; nt < 4; nt++) {
      int col = n0 + wn + nt * 16 + l16;
      float bv = bias[col];
      if (EPI == EPI_VT) {
        // Write V transposed: out[((b*16+h)*128+hd)*2048 + s], 16B chunks.
        int row0 = m0 + wm + mt * 16 + lhalf * 8;
        int b = row0 >> 11;
        int s = row0 & 2047;
        int h = col >> 7;
        int hd = col & 127;
        bf16x8 o;
#pragma unroll
        for (int r = 0; r < 8; r++) o[r] = f2bf(acc[mt][nt][r] + bv);
        long long idx = (((long long)b * 16 + h) * 128 + hd) * 2048 + s;
        *(bf16x8*)((bf16*)outv + idx) = o;
      } else {
#pragma unroll
        for (int r = 0; r < 8; r++) {
          int row = m0 + wm + mt * 16 + r + lhalf * 8;
          long long idx = (long long)row * N + col;
          float v = acc[mt][nt][r] + bv;
          if (EPI == EPI_BF16) {
            ((bf16*)outv)[idx] = f2bf(v);
          } else if (EPI == EPI_RESID_F32) {
            ((float*)outv)[idx] = v + ((const float*)extrav)[idx];
          } else {  // EPI_SILU_MUL: out = silu(gate) * v
            float g = bf2f(((const bf16*)extrav)[idx]);
            float sg = g / (1.f + __expf(-g));
            ((bf16*)outv)[idx] = f2bf(sg * v);
          }
        }
      }
    }
  }
}

// ---------------------------------------------------------------------------
// Flash attention (causal), bf16 WMMA. Block = 4 waves, 64 q rows per block.
// K rows and pre-transposed V rows staged into LDS by contiguous copies.
// ---------------------------------------------------------------------------
__global__ __launch_bounds__(128) void flash_attn_kernel(
    const bf16* __restrict__ Q, const bf16* __restrict__ Km,
    const bf16* __restrict__ Vt_g, bf16* __restrict__ O) {
  constexpr int S = 2048, D = 2048, HD = 128;
  __shared__ __align__(16) bf16 Ks[64][HD];
  __shared__ __align__(16) bf16 Vts[HD][64];
  __shared__ __align__(16) bf16 Ps[4][16][64];
  const int tid = threadIdx.x;
  const int wid = tid >> 5;
  const int lane = tid & 31;
  const int lhalf = lane >> 4;
  const int l16 = lane & 15;
  const int qt = blockIdx.x;
  const int h = blockIdx.y;
  const int b = blockIdx.z;
  const int q0 = qt * 64 + wid * 16;
  const long long rowbase = (long long)b * S * D;
  const long long vtbase = ((long long)b * 16 + h) * HD * S;
  const float scale = 0.08838834764831845f;  // 1/sqrt(128)

  // Q fragments for 4 K-steps of 32 along HD (held for whole KV loop)
  bf16x16 qf[4];
  {
    const bf16* qp = Q + rowbase + (long long)(q0 + l16) * D + h * HD;
    const int a0 = lhalf * 8;
#pragma unroll
    for (int ks = 0; ks < 4; ks++)
      qf[ks] = cat8(*(const bf16x8*)(qp + ks * 32 + a0),
                    *(const bf16x8*)(qp + ks * 32 + a0 + 16));
  }

  f32x8 oacc[8];
#pragma unroll
  for (int j = 0; j < 8; j++) oacc[j] = zero8();
  float m_run[8], l_run[8];
#pragma unroll
  for (int r = 0; r < 8; r++) {
    m_run[r] = -1e30f;
    l_run[r] = 0.f;
  }

  for (int kb = 0; kb <= qt; kb++) {
    const int kv0 = kb * 64;
    // K block rows: [64][128]
    for (int i = tid; i < 64 * (HD / 8); i += 128) {
      int r = i >> 4;
      int c = (i & 15) * 8;
      copy16_to_lds(Km + rowbase + (long long)(kv0 + r) * D + h * HD + c,
                    &Ks[r][c]);
    }
    // V^T block rows: [128][64] from pre-transposed global V
    for (int i = tid; i < HD * (64 / 8); i += 128) {
      int r = i >> 3;
      int c = (i & 7) * 8;
      copy16_to_lds(Vt_g + vtbase + (long long)r * S + kv0 + c, &Vts[r][c]);
    }
    wait_async_lds();
    __syncthreads();

    // Scores: 16 q rows x 64 keys = 4 accumulators, contract over HD=128.
    f32x8 sc[4];
#pragma unroll
    for (int j = 0; j < 4; j++) sc[j] = zero8();
#pragma unroll
    for (int ks = 0; ks < 4; ks++) {
      bf16x16 bfr[4];
      const int b0 = ks * 32 + lhalf * 8;
#pragma unroll
      for (int j = 0; j < 4; j++) {
        const bf16* kp = &Ks[j * 16 + l16][0];
        bfr[j] = cat8(*(const bf16x8*)(kp + b0),
                      *(const bf16x8*)(kp + b0 + 16));
      }
#pragma unroll
      for (int j = 0; j < 4; j++) sc[j] = wmma_bf16(qf[ks], bfr[j], sc[j]);
    }

    // Online softmax per row (rows live in 16-lane halves).
#pragma unroll
    for (int r = 0; r < 8; r++) {
      const int qrow = q0 + r + lhalf * 8;
      float vals[4];
      float mx = -1e30f;
#pragma unroll
      for (int j = 0; j < 4; j++) {
        int key = kv0 + j * 16 + l16;
        float s = sc[j][r] * scale;
        if (key > qrow) s = -1e30f;
        vals[j] = s;
        mx = fmaxf(mx, s);
      }
#pragma unroll
      for (int off = 1; off < 16; off <<= 1)
        mx = fmaxf(mx, __shfl_xor(mx, off, 32));
      float newm = fmaxf(m_run[r], mx);
      float corr = __expf(m_run[r] - newm);
      float rsum = 0.f;
#pragma unroll
      for (int j = 0; j < 4; j++) {
        float p = __expf(vals[j] - newm);
        rsum += p;
        Ps[wid][r + lhalf * 8][j * 16 + l16] = f2bf(p);
      }
#pragma unroll
      for (int off = 1; off < 16; off <<= 1) rsum += __shfl_xor(rsum, off, 32);
      l_run[r] = l_run[r] * corr + rsum;
      m_run[r] = newm;
#pragma unroll
      for (int j2 = 0; j2 < 8; j2++) oacc[j2][r] *= corr;
    }

    // P @ V : contract over 64 keys (2 K-steps of 32), N = HD = 8 tiles.
#pragma unroll
    for (int ks = 0; ks < 2; ks++) {
      const int a0 = ks * 32 + lhalf * 8;
      const bf16* pp = &Ps[wid][l16][0];
      bf16x16 af = cat8(*(const bf16x8*)(pp + a0),
                        *(const bf16x8*)(pp + a0 + 16));
      const int b0 = ks * 32 + lhalf * 8;
      bf16x16 bfv[8];
#pragma unroll
      for (int j2 = 0; j2 < 8; j2++) {
        const bf16* vp = &Vts[j2 * 16 + l16][0];
        bfv[j2] = cat8(*(const bf16x8*)(vp + b0),
                       *(const bf16x8*)(vp + b0 + 16));
      }
#pragma unroll
      for (int j2 = 0; j2 < 8; j2++)
        oacc[j2] = wmma_bf16(af, bfv[j2], oacc[j2]);
    }
    __syncthreads();
  }

  // Normalize and store: O[b*S+q][h*HD + hd] bf16
#pragma unroll
  for (int j2 = 0; j2 < 8; j2++) {
    int col = h * HD + j2 * 16 + l16;
#pragma unroll
    for (int r = 0; r < 8; r++) {
      int qrow = q0 + r + lhalf * 8;
      float v = oacc[j2][r] / l_run[r];
      O[rowbase + (long long)qrow * D + col] = f2bf(v);
    }
  }
}

// ---------------------------------------------------------------------------
// Host launcher
// ---------------------------------------------------------------------------
extern "C" void kernel_launch(void* const* d_in, const int* in_sizes, int n_in,
                              void* d_out, int out_size, void* d_ws,
                              size_t ws_size, hipStream_t stream) {
  (void)in_sizes; (void)n_in; (void)out_size; (void)ws_size;
  const float* x   = (const float*)d_in[0];
  const float* wq  = (const float*)d_in[1];
  const float* bq  = (const float*)d_in[2];
  const float* wk  = (const float*)d_in[3];
  const float* bk  = (const float*)d_in[4];
  const float* wv  = (const float*)d_in[5];
  const float* bv  = (const float*)d_in[6];
  const float* wo  = (const float*)d_in[7];
  const float* bo  = (const float*)d_in[8];
  const float* wg  = (const float*)d_in[9];
  const float* bg  = (const float*)d_in[10];
  const float* wu  = (const float*)d_in[11];
  const float* bu  = (const float*)d_in[12];
  const float* wd  = (const float*)d_in[13];
  const float* bd  = (const float*)d_in[14];
  const float* n1w = (const float*)d_in[15];
  const float* n2w = (const float*)d_in[16];
  float* out = (float*)d_out;

  const long long Bc = 2, Sc = 2048, Dc = 2048;
  const long long Mr = Bc * Sc;  // 4096 rows
  const long long D4 = 4 * Dc;   // 8192

  size_t off = 0;
  auto alloc = [&](size_t bytes) -> void* {
    void* p = (char*)d_ws + off;
    off += (bytes + 255) & ~(size_t)255;
    return p;
  };
  // Transposed bf16 weights: Wt[N][K]
  bf16* wq_t = (bf16*)alloc(Dc * Dc * 2);
  bf16* wk_t = (bf16*)alloc(Dc * Dc * 2);
  bf16* wv_t = (bf16*)alloc(Dc * Dc * 2);
  bf16* wo_t = (bf16*)alloc(Dc * Dc * 2);
  bf16* wg_t = (bf16*)alloc(Dc * D4 * 2);
  bf16* wu_t = (bf16*)alloc(Dc * D4 * 2);
  bf16* wd_t = (bf16*)alloc(D4 * Dc * 2);
  bf16* nx   = (bf16*)alloc(Mr * Dc * 2);
  bf16* qb   = (bf16*)alloc(Mr * Dc * 2);
  bf16* kbuf = (bf16*)alloc(Mr * Dc * 2);
  bf16* vt_g = (bf16*)alloc(Mr * Dc * 2);  // [b][h][hd][S]
  bf16* attn = (bf16*)alloc(Mr * Dc * 2);
  float* x1  = (float*)alloc(Mr * Dc * 4);
  bf16* gate = (bf16*)alloc(Mr * D4 * 2);
  bf16* hbuf = (bf16*)alloc(Mr * D4 * 2);

  // 1) cast + transpose weights: W[K][N] f32 -> Wt[N][K] bf16
  auto castT = [&](const float* src, bf16* dst, long long K, long long N) {
    dim3 g((unsigned)(N / 32), (unsigned)(K / 32));
    cast_transpose_kernel<<<g, 256, 0, stream>>>(src, dst, (int)K, (int)N);
  };
  castT(wq, wq_t, Dc, Dc);
  castT(wk, wk_t, Dc, Dc);
  castT(wv, wv_t, Dc, Dc);
  castT(wo, wo_t, Dc, Dc);
  castT(wg, wg_t, Dc, D4);
  castT(wu, wu_t, Dc, D4);
  castT(wd, wd_t, D4, Dc);

  // 2) rmsnorm(x) -> nx (bf16)
  rmsnorm_kernel<<<(int)Mr, 256, 0, stream>>>(x, n1w, nx, (int)Dc);

  // 3) Q/K/V projections
  dim3 gQ((unsigned)(Dc / BN), (unsigned)(Mr / BM));
  gemm_bf16_kernel<EPI_BF16><<<gQ, 256, 0, stream>>>(
      nx, wq_t, bq, qb, nullptr, (int)Mr, (int)Dc, (int)Dc);
  gemm_bf16_kernel<EPI_BF16><<<gQ, 256, 0, stream>>>(
      nx, wk_t, bk, kbuf, nullptr, (int)Mr, (int)Dc, (int)Dc);
  gemm_bf16_kernel<EPI_VT><<<gQ, 256, 0, stream>>>(
      nx, wv_t, bv, vt_g, nullptr, (int)Mr, (int)Dc, (int)Dc);

  // 4) RoPE on q and k (in place)
  rope_kernel<<<16384, 256, 0, stream>>>(qb);
  rope_kernel<<<16384, 256, 0, stream>>>(kbuf);

  // 5) flash attention -> attn (bf16)
  dim3 gA(32, 16, 2);
  flash_attn_kernel<<<gA, 128, 0, stream>>>(qb, kbuf, vt_g, attn);

  // 6) O projection + residual: x1 = x + attn@wo + bo (f32)
  gemm_bf16_kernel<EPI_RESID_F32><<<gQ, 256, 0, stream>>>(
      attn, wo_t, bo, x1, x, (int)Mr, (int)Dc, (int)Dc);

  // 7) rmsnorm(x1) -> nx (reuse)
  rmsnorm_kernel<<<(int)Mr, 256, 0, stream>>>(x1, n2w, nx, (int)Dc);

  // 8) gate = nx@w_gate + b_gate (bf16)
  dim3 gF((unsigned)(D4 / BN), (unsigned)(Mr / BM));
  gemm_bf16_kernel<EPI_BF16><<<gF, 256, 0, stream>>>(
      nx, wg_t, bg, gate, nullptr, (int)Mr, (int)D4, (int)Dc);

  // 9) h = silu(gate) * (nx@w_up + b_up) (bf16)
  gemm_bf16_kernel<EPI_SILU_MUL><<<gF, 256, 0, stream>>>(
      nx, wu_t, bu, hbuf, gate, (int)Mr, (int)D4, (int)Dc);

  // 10) out = x1 + h@w_down + b_down (f32)
  gemm_bf16_kernel<EPI_RESID_F32><<<gQ, 256, 0, stream>>>(
      hbuf, wd_t, bd, out, x1, (int)Mr, (int)Dc, (int)D4);
}
